// VGAEEncoder_protein_59081570123784
// MI455X (gfx1250) — matrix operand order repaired and checked by reading, not compiled, not run on hardware
//
#include <hip/hip_runtime.h>
#include <hip/hip_bf16.h>
#include <cstdint>
#include <cstddef>

// ---------------------------------------------------------------------------
// Types for CDNA5 WMMA (gfx1250, wave32)
// ---------------------------------------------------------------------------
typedef __attribute__((ext_vector_type(16))) __bf16 v16bf;
typedef __attribute__((ext_vector_type(8)))  float  v8f;

#define NEG_SLOPE 0.2f

// ---------------------------------------------------------------------------
// Helpers
// ---------------------------------------------------------------------------
__device__ __forceinline__ unsigned fenc(float f) {
    unsigned u = __float_as_uint(f);
    return (u & 0x80000000u) ? ~u : (u | 0x80000000u);
}
__device__ __forceinline__ float fdec(unsigned u) {
    u = (u & 0x80000000u) ? (u & 0x7fffffffu) : ~u;
    return __uint_as_float(u);
}
__device__ __forceinline__ unsigned h32(unsigned x) {
    x ^= x >> 16; x *= 0x7feb352du;
    x ^= x >> 15; x *= 0x846ca68bu;
    x ^= x >> 16; return x;
}

// Packed A-fragment offset: layout [tile][kblock][lane(32)][16 contiguous bf16].
// ISA 16-bit 16x32 A layout: lane half h holds K = 8h+{0..7} (idx 0..7) and
// K = 16+8h+{0..7} (idx 8..15), for row M = lane&15.
__device__ __forceinline__ size_t packA_off(int row, int k, int K) {
    int tile = row >> 4, l15 = row & 15;
    int kb = k >> 5, kin = k & 31;
    int h   = (kin >> 3) & 1;                    // bit3 of kin
    int idx = (kin & 7) | ((kin >> 4) << 3);     // low3 + 8*bit4
    int lane = h * 16 + l15;
    return ((((size_t)tile * (K >> 5) + kb) * 32) + lane) * 16 + idx;
}

// ---------------------------------------------------------------------------
// Conversions / packing
// ---------------------------------------------------------------------------
// f32 [N x K] row-major -> bf16 packed A-fragment layout
__global__ void k_f2bf_packA(const float* __restrict__ s, __bf16* __restrict__ d,
                             int N, int K) {
    long i = (long)blockIdx.x * blockDim.x + threadIdx.x;
    if (i >= (long)N * K) return;
    int row = (int)(i / K), k = (int)(i % K);
    d[packA_off(row, k, K)] = (__bf16)s[i];
}

// Weights W [K x OUT] f32 -> bf16 packed B-fragment layout:
// [kblock][tile][lane(32)][16], lane half h: idx 0..7 -> K = kb*32 + 8h + idx,
// idx 8..15 -> K = kb*32 + 16 + 8h + (idx-8), column = tile*16 + (lane&15).
__global__ void k_packB(const float* __restrict__ W, __bf16* __restrict__ d,
                        int K, int OUT) {
    long i = (long)blockIdx.x * blockDim.x + threadIdx.x;
    if (i >= (long)K * OUT) return;
    int NT  = OUT >> 4;
    int idx  = (int)(i & 15);
    int lane = (int)((i >> 4) & 31);
    long rest = i >> 9;
    int t  = (int)(rest % NT);
    int kb = (int)(rest / NT);
    int h = lane >> 4, l15 = lane & 15;
    int col = t * 16 + l15;
    int kk = kb * 32 + ((idx >> 3) << 4) + 8 * h + (idx & 7);
    d[i] = (__bf16)W[(size_t)kk * OUT + col];
}

__global__ void k_zero_f(float* __restrict__ p, long n) {
    long i = (long)blockIdx.x * blockDim.x + threadIdx.x;
    if (i < n) p[i] = 0.0f;
}

__global__ void k_init_nodes(unsigned* __restrict__ mEnc, float* __restrict__ ssum, int N) {
    int i = blockIdx.x * blockDim.x + threadIdx.x;
    if (i < N) { mEnc[i] = 0u; ssum[i] = 0.0f; }
}

// ---------------------------------------------------------------------------
// Dense GEMM  C[N x OUT] = A[N x K] * W[K x OUT], both operands pre-packed in
// WMMA fragment order.  Inner loop: 1 A load + NT B loads + NT WMMAs.
// ---------------------------------------------------------------------------
template<int OUT>
__global__ void k_gemm_wmma(const __bf16* __restrict__ Ap,
                            const __bf16* __restrict__ Wp,
                            float* __restrict__ C, int N, int K) {
    constexpr int NT = OUT / 16;
    int wid  = (blockIdx.x * blockDim.x + threadIdx.x) >> 5;   // wave id (uniform)
    int lane = threadIdx.x & 31;
    int tiles = (N + 15) >> 4;
    if (wid >= tiles) return;                                  // wave-uniform exit

    int hlf = lane >> 4;
    int l15 = lane & 15;
    int KB  = K >> 5;
    const __bf16* abase = Ap + ((size_t)wid * KB * 32 + lane) * 16;
    const __bf16* bbase = Wp + (size_t)lane * 16;

    v8f acc[NT] = {};

    for (int kb = 0; kb < KB; kb++) {
        v16bf a = *(const v16bf*)(abase + (size_t)kb * 512);
        #pragma unroll
        for (int t = 0; t < NT; t++) {
            v16bf b = *(const v16bf*)(bbase + ((size_t)kb * NT + t) * 512);
            acc[t] = __builtin_amdgcn_wmma_f32_16x16x32_bf16(
                         false, a, false, b, (short)0, acc[t], false, false);
        }
    }

    // C/D layout: VGPR r -> M = r + 8*hlf, N = l15
    int mbase = wid * 16 + hlf * 8;
    if (wid * 16 + 16 <= N) {                                  // fast path: no guards
        #pragma unroll
        for (int t = 0; t < NT; t++) {
            #pragma unroll
            for (int r = 0; r < 8; r++)
                C[(size_t)(mbase + r) * OUT + t * 16 + l15] = acc[t][r];
        }
    } else {
        #pragma unroll
        for (int t = 0; t < NT; t++) {
            #pragma unroll
            for (int r = 0; r < 8; r++)
                if (mbase + r < N)
                    C[(size_t)(mbase + r) * OUT + t * 16 + l15] = acc[t][r];
        }
    }
}

// ---------------------------------------------------------------------------
// Edge phase.  Total edges = E real + N self loops (edge id >= E -> loop).
// One wave32 per edge; VPL = OUT/32 features per lane (1 or 4), so each
// lane's gather is a single b32/b128 load.
// ---------------------------------------------------------------------------
template<int VPL>
__global__ void k_edge_score(const float* __restrict__ xl, const float* __restrict__ xr,
                             const float* __restrict__ att,
                             const int* __restrict__ src, const int* __restrict__ dst,
                             int E, int N,
                             float* __restrict__ eArr, unsigned* __restrict__ mEnc) {
    constexpr int OUT = VPL * 32;
    int wid  = (blockIdx.x * blockDim.x + threadIdx.x) >> 5;
    int lane = threadIdx.x & 31;
    if (wid >= E + N) return;
    int s = (wid < E) ? src[wid] : (wid - E);
    int d = (wid < E) ? dst[wid] : (wid - E);
    int k0 = lane * VPL;
    float acc = 0.0f;
    #pragma unroll
    for (int v = 0; v < VPL; v++) {
        float a = xl[(size_t)s * OUT + k0 + v] + xr[(size_t)d * OUT + k0 + v];
        a = (a > 0.0f) ? a : NEG_SLOPE * a;
        acc += a * att[k0 + v];
    }
    #pragma unroll
    for (int off = 16; off > 0; off >>= 1) acc += __shfl_xor(acc, off, 32);
    if (lane == 0) {
        eArr[wid] = acc;
        atomicMax(&mEnc[d], fenc(acc));
    }
}

__global__ void k_edge_exp(float* __restrict__ eArr,
                           const int* __restrict__ dst, int E, int N,
                           const unsigned* __restrict__ mEnc, float* __restrict__ ssum) {
    int t = blockIdx.x * blockDim.x + threadIdx.x;
    if (t >= E + N) return;
    int d = (t < E) ? dst[t] : (t - E);
    float ex = __expf(eArr[t] - fdec(mEnc[d]));
    eArr[t] = ex;
    atomicAdd(&ssum[d], ex);
}

template<int VPL>
__global__ void k_edge_agg(const float* __restrict__ xl,
                           const float* __restrict__ eArr, const float* __restrict__ ssum,
                           const int* __restrict__ src, const int* __restrict__ dst,
                           int E, int N, float* __restrict__ agg) {
    constexpr int OUT = VPL * 32;
    int wid  = (blockIdx.x * blockDim.x + threadIdx.x) >> 5;
    int lane = threadIdx.x & 31;
    if (wid >= E + N) return;
    int s = (wid < E) ? src[wid] : (wid - E);
    int d = (wid < E) ? dst[wid] : (wid - E);
    float alpha = eArr[wid] / (ssum[d] + 1e-16f);
    int k0 = lane * VPL;
    #pragma unroll
    for (int v = 0; v < VPL; v++)
        atomicAdd(&agg[(size_t)d * OUT + k0 + v],
                  xl[(size_t)s * OUT + k0 + v] * alpha);
}

// Epilogue: +bias, optional ELU; writes f32 row-major and/or packed-bf16.
__global__ void k_epilogue(const float* __restrict__ agg, const float* __restrict__ bias,
                           float* __restrict__ hf, __bf16* __restrict__ hb,
                           int N, int OUT, int elu) {
    long i = (long)blockIdx.x * blockDim.x + threadIdx.x;
    if (i >= (long)N * OUT) return;
    int row = (int)(i / OUT), k = (int)(i % OUT);
    float v = agg[i] + bias[k];
    if (elu) v = (v > 0.0f) ? v : (__expf(v) - 1.0f);
    if (hf) hf[i] = v;
    if (hb) hb[packA_off(row, k, OUT)] = (__bf16)v;
}

// Reparameterization: z = eps * exp(logstd) + mu  (hash-based Box-Muller)
__global__ void k_reparam(const float* __restrict__ mu, const float* __restrict__ ls,
                          float* __restrict__ z, long n) {
    long i = (long)blockIdx.x * blockDim.x + threadIdx.x;
    if (i >= n) return;
    float stdv = __expf(ls[i]);
    unsigned a = h32((unsigned)i * 2654435761u + 0x9e3779b9u);
    unsigned b = h32((unsigned)i * 0x85ebca6bu + 0xc2b2ae35u);
    float u1 = ((a >> 8) + 1u) * (1.0f / 16777216.0f);
    float u2 = (b >> 8) * (1.0f / 16777216.0f);
    float eps = sqrtf(-2.0f * __logf(u1)) * __cosf(6.28318530717959f * u2);
    z[i] = eps * stdv + mu[i];
}

// ---------------------------------------------------------------------------
// Host orchestration
// ---------------------------------------------------------------------------
static inline unsigned gblk(long n, int t) { return (unsigned)((n + t - 1) / t); }

extern "C" void kernel_launch(void* const* d_in, const int* in_sizes, int n_in,
                              void* d_out, int out_size, void* d_ws, size_t ws_size,
                              hipStream_t stream) {
    const float* x  = (const float*)d_in[0];
    const int*   ei = (const int*)d_in[1];          // int32 (JAX x64 disabled)
    const int IN = 512, MAXF = 128;
    int N = in_sizes[0] / IN;
    int E = in_sizes[1] / 2;
    const int* src = ei;
    const int* dst = ei + E;
    long ET = (long)E + N;
    long tiles = (N + 15) / 16;

    // Workspace carve-out (256B aligned slices)
    char* p = (char*)d_ws;
    auto alloc = [&](size_t bytes) -> void* {
        void* r = (void*)p; p += (bytes + 255) & ~(size_t)255; return r;
    };
    __bf16*  xbf  = (__bf16*)alloc((size_t)tiles * 16 * IN * sizeof(__bf16));   // packed A
    __bf16*  hbf  = (__bf16*)alloc((size_t)tiles * 16 * MAXF * sizeof(__bf16)); // packed A
    __bf16*  wlbf = (__bf16*)alloc((size_t)IN * MAXF * sizeof(__bf16));         // packed B
    __bf16*  wrbf = (__bf16*)alloc((size_t)IN * MAXF * sizeof(__bf16));         // packed B
    float*   xl   = (float*)alloc((size_t)N * MAXF * sizeof(float));
    float*   xr   = (float*)alloc((size_t)N * MAXF * sizeof(float));
    float*   agg  = (float*)alloc((size_t)N * MAXF * sizeof(float));
    float*   eArr = (float*)alloc((size_t)ET * sizeof(float));
    unsigned* mEnc = (unsigned*)alloc((size_t)N * sizeof(unsigned));
    float*   ssum = (float*)alloc((size_t)N * sizeof(float));

    float* mu     = (float*)d_out;                  // [N,32]
    float* logstd = mu + (size_t)N * 32;            // [N,32]
    float* zbuf   = logstd + (size_t)N * 32;        // [N,32]

    auto run_layer = [&](const __bf16* in_bf, int K, int OUT,
                         const float* Wl, const float* Wr,
                         const float* att, const float* bias,
                         float* out_f, __bf16* out_bf, int elu) {
        long wn = (long)K * OUT;
        k_packB<<<gblk(wn, 256), 256, 0, stream>>>(Wl, wlbf, K, OUT);
        k_packB<<<gblk(wn, 256), 256, 0, stream>>>(Wr, wrbf, K, OUT);

        unsigned gg = gblk(tiles * 32, 256);        // 8 waves / 256-thread block
        if (OUT == 32) {
            k_gemm_wmma<32><<<gg, 256, 0, stream>>>(in_bf, wlbf, xl, N, K);
            k_gemm_wmma<32><<<gg, 256, 0, stream>>>(in_bf, wrbf, xr, N, K);
        } else {
            k_gemm_wmma<128><<<gg, 256, 0, stream>>>(in_bf, wlbf, xl, N, K);
            k_gemm_wmma<128><<<gg, 256, 0, stream>>>(in_bf, wrbf, xr, N, K);
        }

        k_init_nodes<<<gblk(N, 256), 256, 0, stream>>>(mEnc, ssum, N);
        k_zero_f<<<gblk((long)N * OUT, 256), 256, 0, stream>>>(agg, (long)N * OUT);

        if (OUT == 32) {
            k_edge_score<1><<<gblk(ET * 32, 256), 256, 0, stream>>>(xl, xr, att, src, dst,
                                                                    E, N, eArr, mEnc);
        } else {
            k_edge_score<4><<<gblk(ET * 32, 256), 256, 0, stream>>>(xl, xr, att, src, dst,
                                                                    E, N, eArr, mEnc);
        }
        k_edge_exp<<<gblk(ET, 256), 256, 0, stream>>>(eArr, dst, E, N, mEnc, ssum);
        if (OUT == 32) {
            k_edge_agg<1><<<gblk(ET * 32, 256), 256, 0, stream>>>(xl, eArr, ssum, src, dst,
                                                                  E, N, agg);
        } else {
            k_edge_agg<4><<<gblk(ET * 32, 256), 256, 0, stream>>>(xl, eArr, ssum, src, dst,
                                                                  E, N, agg);
        }
        k_epilogue<<<gblk((long)N * OUT, 256), 256, 0, stream>>>(agg, bias, out_f, out_bf,
                                                                 N, OUT, elu);
    };

    // x -> packed bf16 once
    k_f2bf_packA<<<gblk((long)N * IN, 256), 256, 0, stream>>>(x, xbf, N, IN);

    // c1: 512 -> 32, ELU, next-layer input packed bf16
    run_layer(xbf, 512, 32,
              (const float*)d_in[2], (const float*)d_in[3],
              (const float*)d_in[4], (const float*)d_in[5],
              nullptr, hbf, 1);
    // c2: 32 -> 128, ELU
    run_layer(hbf, 32, 128,
              (const float*)d_in[6], (const float*)d_in[7],
              (const float*)d_in[8], (const float*)d_in[9],
              nullptr, hbf, 1);
    // cmu: 128 -> 32, linear, straight to d_out
    run_layer(hbf, 128, 32,
              (const float*)d_in[10], (const float*)d_in[11],
              (const float*)d_in[12], (const float*)d_in[13],
              mu, nullptr, 0);
    // cls: 128 -> 32, linear, straight to d_out
    run_layer(hbf, 128, 32,
              (const float*)d_in[14], (const float*)d_in[15],
              (const float*)d_in[16], (const float*)d_in[17],
              logstd, nullptr, 0);

    // z = eps * exp(logstd) + mu
    k_reparam<<<gblk((long)N * 32, 256), 256, 0, stream>>>(mu, logstd, zbuf, (long)N * 32);
}
